// DeformableTransformerEncoderLayer_52913997087413
// MI455X (gfx1250) — matrix operand order repaired and checked by reading, not compiled, not run on hardware
//
#include <hip/hip_runtime.h>
#include <math.h>

// ---------------- problem constants (from reference) ----------------
#define NTOK   13294
#define NBATCH 2
#define NTOT   (NBATCH * NTOK)          // 26588
#define CDIM   256
#define NHEAD  8
#define HDIM   32
#define NLVL   4
#define NPTS   4
#define DFFN   1024

typedef unsigned short u16;
typedef __attribute__((ext_vector_type(16))) __bf16 bf16x16;   // 32B: WMMA A/B frag
typedef __attribute__((ext_vector_type(8)))  float  floatx8;   // 32B: WMMA C/D frag

struct Pack32 { uint4 lo; uint4 hi; };

static __device__ __forceinline__ bf16x16 pack_frag(uint4 lo, uint4 hi) {
    Pack32 t; t.lo = lo; t.hi = hi;
    return __builtin_bit_cast(bf16x16, t);
}

// f32 -> bf16 round-to-nearest-even (bit trick; no __bf16 arithmetic needed)
static __device__ __forceinline__ u16 f2bf(float f) {
    unsigned int u = __float_as_uint(f);
    unsigned int r = u + 0x7fffu + ((u >> 16) & 1u);
    return (u16)(r >> 16);
}

// ---------------------------------------------------------------------------
// Generic bf16 WMMA GEMM:  C[M x N] = A[M x K] * W[N x K]^T + bias
// Block: 256 threads = 8 waves; each wave -> 16 rows x 64 cols.
// blockIdx.x -> 64-col tile, blockIdx.y -> 128-row tile.
// ---------------------------------------------------------------------------
template<int RELU, int STORE_BF16>
__global__ __launch_bounds__(256)
void gemm_wmma_bf16(const u16* __restrict__ A, const u16* __restrict__ W,
                    const float* __restrict__ bias,
                    float* __restrict__ Cf, u16* __restrict__ Cb,
                    int Mrows, int N, int K)
{
    const int lane    = threadIdx.x & 31;
    const int wave    = threadIdx.x >> 5;
    const int rsel    = lane & 15;       // row (A) / col (B) / col (D) selector
    const int khalf   = lane >> 4;       // which K half this lane covers
    const int rowBase = blockIdx.y * 128 + wave * 16;
    const int colBase = blockIdx.x * 64;

    int rowA = rowBase + rsel;
    if (rowA >= Mrows) rowA = Mrows - 1;         // clamp edge reads, mask stores

    // A (16x32 bf16): lane holds row rsel; chunks at K = khalf*8 and 16+khalf*8
    const u16* aRow = A + (size_t)rowA * K + khalf * 8;
    // B (32x16 bf16): lane holds column; chunks at K = khalf*8 and 16+khalf*8
    const u16* wCol[4];
#pragma unroll
    for (int j = 0; j < 4; ++j)
        wCol[j] = W + (size_t)(colBase + j * 16 + rsel) * K + khalf * 8;

    floatx8 acc[4] = { {}, {}, {}, {} };

    const int ksteps = K >> 5;
    for (int kk = 0; kk < ksteps; ++kk) {
        const u16* ap = aRow + kk * 32;
        bf16x16 afrag = pack_frag(*(const uint4*)ap, *(const uint4*)(ap + 16));
#pragma unroll
        for (int j = 0; j < 4; ++j) {
            const u16* bp = wCol[j] + kk * 32;
            bf16x16 bfrag = pack_frag(*(const uint4*)bp, *(const uint4*)(bp + 16));
            acc[j] = __builtin_amdgcn_wmma_f32_16x16x32_bf16(
                         false, afrag, false, bfrag, (short)0, acc[j], false, false);
        }
    }

    // Epilogue: D layout — VGPR v, lane -> row v + 8*khalf, col rsel
#pragma unroll
    for (int j = 0; j < 4; ++j) {
        const int col = colBase + j * 16 + rsel;
        const float bv = bias[col];
#pragma unroll
        for (int v = 0; v < 8; ++v) {
            const int r = rowBase + 8 * khalf + v;
            if (r < Mrows) {
                float val = acc[j][v] + bv;
                if (RELU) val = fmaxf(val, 0.f);
                if (STORE_BF16) Cb[(size_t)r * N + col] = f2bf(val);
                else            Cf[(size_t)r * N + col] = val;
            }
        }
    }
}

// ---------------------------------------------------------------------------
// LSQ fake-quant (clip/round/rescale) + f32->bf16 weight conversion
// ---------------------------------------------------------------------------
__global__ __launch_bounds__(256)
void quant_to_bf16(const float* __restrict__ w, const float* __restrict__ alpha,
                   u16* __restrict__ out, int n, int doq)
{
    int i = blockIdx.x * blockDim.x + threadIdx.x;
    if (i >= n) return;
    float v = w[i];
    if (doq) {
        float a  = alpha[0];
        float wc = fminf(fmaxf(v / a, -8.f), 7.f);
        v = rintf(wc) * a;                 // round-half-even, matches jnp.round
    }
    out[i] = f2bf(v);
}

__global__ __launch_bounds__(256)
void add_pos_cvt(const float* __restrict__ src, const float* __restrict__ pos,
                 u16* __restrict__ qb, u16* __restrict__ sb, int n)
{
    int i = blockIdx.x * blockDim.x + threadIdx.x;
    if (i >= n) return;
    float s = src[i];
    qb[i] = f2bf(s + pos[i]);
    sb[i] = f2bf(s);
}

// ---------------------------------------------------------------------------
// Deformable attention core: one thread per (token, head).
// softmax over 16 (L*P) logits, bilinear zero-padded sampling from value f32,
// accumulate 32 channels, write bf16 row for the out-proj GEMM.
// ---------------------------------------------------------------------------
__global__ __launch_bounds__(256)
void deform_attn(const float* __restrict__ value,
                 const float* __restrict__ sampoff,
                 const float* __restrict__ logits,
                 const float* __restrict__ refp,
                 u16* __restrict__ attn_out)
{
    int gid = blockIdx.x * blockDim.x + threadIdx.x;
    if (gid >= NTOT * NHEAD) return;
    const int tok = gid >> 3;
    const int m   = gid & 7;
    const int b   = tok / NTOK;

    // softmax over the 16 (level,point) weights of this head
    float w[16];
    const float* lp = logits + (size_t)tok * (NHEAD * 16) + m * 16;
    float mx = -3.0e38f;
#pragma unroll
    for (int j = 0; j < 16; ++j) { w[j] = lp[j]; mx = fmaxf(mx, w[j]); }
    float s = 0.f;
#pragma unroll
    for (int j = 0; j < 16; ++j) { w[j] = expf(w[j] - mx); s += w[j]; }
    const float inv = 1.f / s;

    float4 acc[8];
#pragma unroll
    for (int d = 0; d < 8; ++d) acc[d] = make_float4(0.f, 0.f, 0.f, 0.f);

    const int Hs[4] = {100, 50, 25, 13};
    const int S0[4] = {0, 10000, 12500, 13125};
    const float* op = sampoff + (size_t)tok * CDIM + m * 32;
    const float* rp = refp + (size_t)tok * (NLVL * 2);

#pragma unroll
    for (int l = 0; l < 4; ++l) {
        const int Hl = Hs[l], Wl = Hs[l], s0 = S0[l];    // square levels
        const float fW = (float)Wl, fH = (float)Hl;
        const float rx = rp[l * 2 + 0], ry = rp[l * 2 + 1];
#pragma unroll
        for (int p = 0; p < 4; ++p) {
            const float ox = op[l * 8 + p * 2 + 0];
            const float oy = op[l * 8 + p * 2 + 1];
            const float x = (rx + ox / fW) * fW - 0.5f;
            const float y = (ry + oy / fH) * fH - 0.5f;
            const float x0f = floorf(x), y0f = floorf(y);
            const int   x0 = (int)x0f,  y0 = (int)y0f;
            const float wx1 = x - x0f, wx0 = 1.f - wx1;
            const float wy1 = y - y0f, wy0 = 1.f - wy1;
            const float aw = w[l * 4 + p] * inv;
#pragma unroll
            for (int cy = 0; cy < 2; ++cy) {
#pragma unroll
                for (int cx = 0; cx < 2; ++cx) {
                    const int xi = x0 + cx, yi = y0 + cy;
                    if (xi < 0 || xi >= Wl || yi < 0 || yi >= Hl) continue;
                    const float cw = aw * (cx ? wx1 : wx0) * (cy ? wy1 : wy0);
                    const float4* vp = (const float4*)(value +
                        ((size_t)(b * NTOK + s0 + yi * Wl + xi)) * CDIM + m * 32);
#pragma unroll
                    for (int d = 0; d < 8; ++d) {
                        float4 v4 = vp[d];
                        acc[d].x += cw * v4.x; acc[d].y += cw * v4.y;
                        acc[d].z += cw * v4.z; acc[d].w += cw * v4.w;
                    }
                }
            }
        }
    }

    u16* o = attn_out + (size_t)tok * CDIM + m * 32;
#pragma unroll
    for (int d = 0; d < 8; ++d) {
        o[4 * d + 0] = f2bf(acc[d].x); o[4 * d + 1] = f2bf(acc[d].y);
        o[4 * d + 2] = f2bf(acc[d].z); o[4 * d + 3] = f2bf(acc[d].w);
    }
}

// ---------------------------------------------------------------------------
// LayerNorm(a + r): one wave32 per 256-channel token row; 8 channels/lane,
// wave reduction via __shfl_xor. Optionally emits bf16 copy for next GEMM.
// ---------------------------------------------------------------------------
template<int EMIT_BF16>
__global__ __launch_bounds__(256)
void ln_residual(const float* __restrict__ a, const float* __restrict__ r,
                 const float* __restrict__ g, const float* __restrict__ be,
                 float* __restrict__ outf, u16* __restrict__ outb)
{
    const int lane = threadIdx.x & 31;
    const int wave = threadIdx.x >> 5;
    const int tok  = blockIdx.x * 8 + wave;
    if (tok >= NTOT) return;

    const float4* ap = (const float4*)(a + (size_t)tok * CDIM) + lane * 2;
    const float4* rp = (const float4*)(r + (size_t)tok * CDIM) + lane * 2;
    float4 v0 = ap[0], v1 = ap[1];
    float4 u0 = rp[0], u1 = rp[1];
    v0.x += u0.x; v0.y += u0.y; v0.z += u0.z; v0.w += u0.w;
    v1.x += u1.x; v1.y += u1.y; v1.z += u1.z; v1.w += u1.w;

    float sum = v0.x + v0.y + v0.z + v0.w + v1.x + v1.y + v1.z + v1.w;
    float sq  = v0.x*v0.x + v0.y*v0.y + v0.z*v0.z + v0.w*v0.w
              + v1.x*v1.x + v1.y*v1.y + v1.z*v1.z + v1.w*v1.w;
#pragma unroll
    for (int mlane = 1; mlane < 32; mlane <<= 1) {
        sum += __shfl_xor(sum, mlane, 32);
        sq  += __shfl_xor(sq,  mlane, 32);
    }
    const float mu  = sum * (1.f / 256.f);
    const float var = sq * (1.f / 256.f) - mu * mu;
    const float inv = rsqrtf(var + 1e-5f);

    const float4* gp = (const float4*)g  + lane * 2;
    const float4* bp = (const float4*)be + lane * 2;
    float4 g0 = gp[0], g1 = gp[1], b0 = bp[0], b1 = bp[1];
    float4 o0, o1;
    o0.x = (v0.x - mu) * inv * g0.x + b0.x;  o0.y = (v0.y - mu) * inv * g0.y + b0.y;
    o0.z = (v0.z - mu) * inv * g0.z + b0.z;  o0.w = (v0.w - mu) * inv * g0.w + b0.w;
    o1.x = (v1.x - mu) * inv * g1.x + b1.x;  o1.y = (v1.y - mu) * inv * g1.y + b1.y;
    o1.z = (v1.z - mu) * inv * g1.z + b1.z;  o1.w = (v1.w - mu) * inv * g1.w + b1.w;

    float4* of = (float4*)(outf + (size_t)tok * CDIM) + lane * 2;
    of[0] = o0; of[1] = o1;
    if (EMIT_BF16) {
        u16* ob = outb + (size_t)tok * CDIM + lane * 8;
        ob[0] = f2bf(o0.x); ob[1] = f2bf(o0.y); ob[2] = f2bf(o0.z); ob[3] = f2bf(o0.w);
        ob[4] = f2bf(o1.x); ob[5] = f2bf(o1.y); ob[6] = f2bf(o1.z); ob[7] = f2bf(o1.w);
    }
}

// ---------------------------------------------------------------------------
extern "C" void kernel_launch(void* const* d_in, const int* in_sizes, int n_in,
                              void* d_out, int out_size, void* d_ws, size_t ws_size,
                              hipStream_t stream) {
    (void)in_sizes; (void)n_in; (void)out_size; (void)ws_size;

    const float* src     = (const float*)d_in[0];
    const float* pos     = (const float*)d_in[1];
    const float* refp    = (const float*)d_in[2];
    const float* vproj_w = (const float*)d_in[3];
    const float* vproj_b = (const float*)d_in[4];
    const float* alpha_v = (const float*)d_in[5];
    const float* samp_w  = (const float*)d_in[6];
    const float* samp_b  = (const float*)d_in[7];
    const float* attn_w  = (const float*)d_in[8];
    const float* attn_b  = (const float*)d_in[9];
    const float* oproj_w = (const float*)d_in[10];
    const float* oproj_b = (const float*)d_in[11];
    const float* alpha_o = (const float*)d_in[12];
    const float* ln1_g   = (const float*)d_in[13];
    const float* ln1_b   = (const float*)d_in[14];
    const float* lin1_w  = (const float*)d_in[15];
    const float* lin1_b  = (const float*)d_in[16];
    const float* alpha1  = (const float*)d_in[17];
    const float* lin2_w  = (const float*)d_in[18];
    const float* lin2_b  = (const float*)d_in[19];
    const float* alpha2  = (const float*)d_in[20];
    const float* ln2_g   = (const float*)d_in[21];
    const float* ln2_b   = (const float*)d_in[22];
    // d_in[23]=spatial_shapes, d_in[24]=level_start_index: compile-time constants

    // ---- workspace layout (with aliasing reuse) ----
    const size_t N256h = (size_t)NTOT * CDIM * 2;   // bf16 [NTOT][256]
    const size_t N256f = (size_t)NTOT * CDIM * 4;   // f32  [NTOT][256]
    const size_t N128f = (size_t)NTOT * 128  * 4;   // f32  [NTOT][128]
    char* w = (char*)d_ws;
    u16*   q_b    = (u16*)  (w);                                  // q = src+pos (bf16)
    u16*   src_b  = (u16*)  (w + N256h);                          // src (bf16)
    float* valf   = (float*)(w + 2 * N256h);                      // value (f32)
    float* sampf  = (float*)(w + 2 * N256h + N256f);              // sampling offsets
    float* logf   = (float*)(w + 2 * N256h + 2 * N256f);          // attn logits
    u16*   aout_b = (u16*)  (w + 2 * N256h + 2 * N256f + N128f);  // attn out (bf16)
    float* src2f  = (float*)(w + 3 * N256h + 2 * N256f + N128f);  // out-proj result
    float* xf     = (float*)(w + 3 * N256h + 3 * N256f + N128f);  // LN1 out (f32)
    u16*   x_b    = (u16*)  (w + 3 * N256h + 4 * N256f + N128f);  // LN1 out (bf16)
    const size_t WOFF = 4 * N256h + 4 * N256f + N128f;
    u16*   wq_v = (u16*)(w + WOFF);               // 256*256 quantized value_proj
    u16*   wq_o = wq_v + 256 * 256;               // 256*256 quantized out_proj
    u16*   wb_s = wq_o + 256 * 256;               // 256*256 samp_w
    u16*   wb_a = wb_s + 256 * 256;               // 128*256 attn_w
    u16*   wq_1 = wb_a + 128 * 256;               // 1024*256 quantized lin1
    u16*   wq_2 = wq_1 + 1024 * 256;              // 256*1024 quantized lin2
    // aliases over dead regions:
    u16*   h_b  = (u16*)(w);                      // FFN hidden bf16 [NTOT][1024]
                                                  //   == q_b+src_b+valf exactly
    float* h2f  = sampf;                          // FFN2 result f32 [NTOT][256]

    const int  NE  = NTOT * CDIM;
    const dim3 blk(256);
    const dim3 gR((unsigned)((NTOT + 127) / 128));   // 208 row supertiles

    // ---- weight prep ----
    quant_to_bf16<<<(256 * 256 + 255) / 256, blk, 0, stream>>>(vproj_w, alpha_v, wq_v, 256 * 256, 1);
    quant_to_bf16<<<(256 * 256 + 255) / 256, blk, 0, stream>>>(oproj_w, alpha_o, wq_o, 256 * 256, 1);
    quant_to_bf16<<<(256 * 256 + 255) / 256, blk, 0, stream>>>(samp_w, nullptr, wb_s, 256 * 256, 0);
    quant_to_bf16<<<(128 * 256 + 255) / 256, blk, 0, stream>>>(attn_w, nullptr, wb_a, 128 * 256, 0);
    quant_to_bf16<<<(1024 * 256 + 255) / 256, blk, 0, stream>>>(lin1_w, alpha1, wq_1, 1024 * 256, 1);
    quant_to_bf16<<<(256 * 1024 + 255) / 256, blk, 0, stream>>>(lin2_w, alpha2, wq_2, 256 * 1024, 1);

    // ---- q = src + pos (bf16), src (bf16) ----
    add_pos_cvt<<<(NE + 255) / 256, blk, 0, stream>>>(src, pos, q_b, src_b, NE);

    // ---- projections (WMMA) ----
    gemm_wmma_bf16<0, 0><<<dim3(4, gR.x), blk, 0, stream>>>(src_b, wq_v, vproj_b, valf, nullptr, NTOT, 256, 256);
    gemm_wmma_bf16<0, 0><<<dim3(4, gR.x), blk, 0, stream>>>(q_b, wb_s, samp_b, sampf, nullptr, NTOT, 256, 256);
    gemm_wmma_bf16<0, 0><<<dim3(2, gR.x), blk, 0, stream>>>(q_b, wb_a, attn_b, logf, nullptr, NTOT, 128, 256);

    // ---- deformable attention gather ----
    deform_attn<<<(NTOT * NHEAD + 255) / 256, blk, 0, stream>>>(valf, sampf, logf, refp, aout_b);

    // ---- out projection + LN1 ----
    gemm_wmma_bf16<0, 0><<<dim3(4, gR.x), blk, 0, stream>>>(aout_b, wq_o, oproj_b, src2f, nullptr, NTOT, 256, 256);
    ln_residual<1><<<(NTOT + 7) / 8, blk, 0, stream>>>(src, src2f, ln1_g, ln1_b, xf, x_b);

    // ---- FFN (WMMA) + LN2 ----
    gemm_wmma_bf16<1, 1><<<dim3(16, gR.x), blk, 0, stream>>>(x_b, wq_1, lin1_b, nullptr, h_b, NTOT, 1024, 256);
    gemm_wmma_bf16<0, 0><<<dim3(4, gR.x), blk, 0, stream>>>(h_b, wq_2, lin2_b, h2f, nullptr, NTOT, 256, 1024);
    ln_residual<0><<<(NTOT + 7) / 8, blk, 0, stream>>>(xf, h2f, ln2_g, ln2_b, (float*)d_out, nullptr);
}
